// LSTM_ABR_18751827214418
// MI455X (gfx1250) — matrix operand order repaired and checked by reading, not compiled, not run on hardware
//
#include <hip/hip_runtime.h>
#include <cstdint>

typedef __attribute__((ext_vector_type(16))) _Float16 v16h;
typedef __attribute__((ext_vector_type(8)))  _Float16 v8h;
typedef __attribute__((ext_vector_type(8)))  float    v8f;

#define HSZ 64
#define TT  32

__device__ __forceinline__ float fsigmoid(float x) { return 1.0f / (1.0f + __expf(-x)); }
__device__ __forceinline__ float ftanhf(float x)   { return 2.0f * fsigmoid(2.0f * x) - 1.0f; }

__device__ __forceinline__ v8f wmma_f16(v16h a, v16h b, v8f c) {
    return __builtin_amdgcn_wmma_f32_16x16x32_f16(false, a, false, b, (short)0, c, false, false);
}

// A operand (16x32 f16) for lane `lane`, K-chunk kc, from row-major [16][64] f16 buffer.
// Lane layout: m = lane&15, half = lane>>4; e<8 -> K = kc*32 + 8*half + e;
//              e>=8 -> K = kc*32 + 16 + 8*half + (e-8).
__device__ __forceinline__ v16h load_A16x64(const _Float16* hs, int lane, int kc) {
    const int m = lane & 15, half = lane >> 4;
    const int base = m * 64 + kc * 32 + 8 * half;
    v8h lo = *(const v8h*)(hs + base);
    v8h hi = *(const v8h*)(hs + base + 16);
    v16h a;
#pragma unroll
    for (int e = 0; e < 8; ++e) { a[e] = lo[e]; a[8 + e] = hi[e]; }
    return a;
}

// B operand (32x16 f16) from row-major weight [N][ld] f16 (B[k][n] = w[n][k]).
// Lane layout: n = n_base + (lane&15); e<8 -> K_local = 8*(lane>>4) + e; e>=8 -> K_local = 16 + 8*(lane>>4) + (e-8).
__device__ __forceinline__ v16h load_B(const _Float16* w, int lane, int n_base, int kc, int ld) {
    const int n = n_base + (lane & 15), half = lane >> 4;
    const int base = n * ld + kc * 32 + 8 * half;
    v8h lo = *(const v8h*)(w + base);
    v8h hi = *(const v8h*)(w + base + 16);
    v16h b;
#pragma unroll
    for (int e = 0; e < 8; ++e) { b[e] = lo[e]; b[8 + e] = hi[e]; }
    return b;
}

struct LstmArgs {
    const float* x[6];
    const float* wih[6];
    const float* whh[6];
    const float* b[6];
    float* hn;   // [6][B][64]
    int B;
};

__global__ __launch_bounds__(256) void lstm6_kernel(LstmArgs args) {
    __shared__ __attribute__((aligned(32))) _Float16 s_whh[256 * 64];   // 32 KB
    __shared__ __attribute__((aligned(16))) float    s_x[8][16 * 32];   // 16 KB
    __shared__ __attribute__((aligned(32))) _Float16 s_h[8][16 * 64];   // 16 KB

    const int run  = blockIdx.y;
    const int tid  = threadIdx.x;
    const int w    = tid >> 5;
    const int lane = tid & 31;
    const int half = lane >> 4;
    const int nlo  = lane & 15;
    const int B    = args.B;
    const int bb   = (blockIdx.x * 8 + w) * 16;

    // Stage whh (f32 -> f16) into LDS cooperatively.
    const float* whh = args.whh[run];
#pragma unroll
    for (int i = 0; i < 64; ++i) {
        const int idx = tid + i * 256;
        s_whh[idx] = (_Float16)whh[idx];
    }

    // Stage this wave's x tile [16 rows][32 steps] and zero the h scratch.
    const float* x = args.x[run];
    for (int i = lane; i < 512; i += 32)
        s_x[w][i] = x[(size_t)(bb + (i >> 5)) * TT + (i & 31)];
    for (int i = lane; i < 512; i += 32)
        ((uint32_t*)s_h[w])[i] = 0u;   // 1024 halves

    __syncthreads();

    // Per-lane input-gate weights/bias (one gate column per lane per tile).
    const float* wih  = args.wih[run];
    const float* bias = args.b[run];
    float wihv[16], bv[16];
#pragma unroll
    for (int nt = 0; nt < 16; ++nt) { wihv[nt] = wih[nt * 16 + nlo]; bv[nt] = bias[nt * 16 + nlo]; }

    v8f c[4], hlast[4];
#pragma unroll
    for (int k = 0; k < 4; ++k)
#pragma unroll
        for (int j = 0; j < 8; ++j) { c[k][j] = 0.0f; hlast[k][j] = 0.0f; }

    for (int t = 0; t < TT; ++t) {
        float xr[8];
#pragma unroll
        for (int j = 0; j < 8; ++j) xr[j] = s_x[w][(j + 8 * half) * TT + t];

        // Read previous h (A operand layout) BEFORE this step's h writes.
        v16h A0 = load_A16x64(s_h[w], lane, 0);
        v16h A1 = load_A16x64(s_h[w], lane, 1);

#pragma unroll
        for (int k = 0; k < 4; ++k) {   // hidden-column tile k: gates i,f,g,o tiles k,4+k,8+k,12+k
            v8f gi, gf, gg, go;
#pragma unroll
            for (int j = 0; j < 8; ++j) {
                gi[j] = xr[j] * wihv[k]      + bv[k];
                gf[j] = xr[j] * wihv[4 + k]  + bv[4 + k];
                gg[j] = xr[j] * wihv[8 + k]  + bv[8 + k];
                go[j] = xr[j] * wihv[12 + k] + bv[12 + k];
            }
            gi = wmma_f16(A0, load_B(s_whh, lane, (k) * 16,      0, 64), gi);
            gi = wmma_f16(A1, load_B(s_whh, lane, (k) * 16,      1, 64), gi);
            gf = wmma_f16(A0, load_B(s_whh, lane, (4 + k) * 16,  0, 64), gf);
            gf = wmma_f16(A1, load_B(s_whh, lane, (4 + k) * 16,  1, 64), gf);
            gg = wmma_f16(A0, load_B(s_whh, lane, (8 + k) * 16,  0, 64), gg);
            gg = wmma_f16(A1, load_B(s_whh, lane, (8 + k) * 16,  1, 64), gg);
            go = wmma_f16(A0, load_B(s_whh, lane, (12 + k) * 16, 0, 64), go);
            go = wmma_f16(A1, load_B(s_whh, lane, (12 + k) * 16, 1, 64), go);

#pragma unroll
            for (int j = 0; j < 8; ++j) {
                const float cn = fsigmoid(gf[j]) * c[k][j] + fsigmoid(gi[j]) * ftanhf(gg[j]);
                c[k][j] = cn;
                const float hv = fsigmoid(go[j]) * ftanhf(cn);
                hlast[k][j] = hv;
                s_h[w][(j + 8 * half) * 64 + k * 16 + nlo] = (_Float16)hv;
            }
        }
    }

    // Write final hidden state (f32) to workspace: hn[run][B][64].
#pragma unroll
    for (int k = 0; k < 4; ++k)
#pragma unroll
        for (int j = 0; j < 8; ++j)
            args.hn[((size_t)run * B + bb + j + 8 * half) * 64 + k * 16 + nlo] = hlast[k][j];
}

struct HeadArgs {
    const float* s2;
    const float* hn;               // [6][B][64]
    const float* w3[2]; const float* b3[2];   // q03 / i03 : [64][3], [64]
    const float* w1[2]; const float* b1[2];   // q1  / i1  : [64][256], [64]
    const float* w2[2]; const float* b2[2];   // q2  / i2  : [6][64], [6]
    float* out;
    int B;
};

__global__ __launch_bounds__(256) void head_kernel(HeadArgs args) {
    __shared__ __attribute__((aligned(32))) _Float16 s_w1[64 * 256];   // 32 KB
    __shared__ __attribute__((aligned(32))) _Float16 s_mid[8][16 * 64];// 16 KB
    __shared__ float s_i6[8][16 * 8];                                  // 4 KB

    const int tid  = threadIdx.x;
    const int w    = tid >> 5;
    const int lane = tid & 31;
    const int half = lane >> 4;
    const int nlo  = lane & 15;
    const int B    = args.B;
    const int bb   = (blockIdx.x * 8 + w) * 16;
    const int arow = bb + nlo;   // A-operand row handled by this lane

    float s2v[3];
#pragma unroll
    for (int d = 0; d < 3; ++d) s2v[d] = args.s2[(size_t)arow * 3 + d];

    for (int br = 0; br < 2; ++br) {   // 0 = q branch, 1 = i branch
        __syncthreads();
        const float* w1 = args.w1[br];
#pragma unroll
        for (int i = 0; i < 64; ++i) s_w1[tid + i * 256] = (_Float16)w1[tid + i * 256];
        __syncthreads();

        // Build A operand chunks for hq = [hn0 | hn1 | hn2 | relu(s2@w3.T+b3)].
        v16h A[8];
        const float* hnb = args.hn + (size_t)(br * 3) * B * 64;
#pragma unroll
        for (int kc = 0; kc < 6; ++kc) {
            const float* base = hnb + (size_t)(kc >> 1) * B * 64
                                    + (size_t)arow * 64 + (kc & 1) * 32 + 8 * half;
            v8f g0 = *(const v8f*)(base);
            v8f g1 = *(const v8f*)(base + 16);
#pragma unroll
            for (int e = 0; e < 8; ++e) { A[kc][e] = (_Float16)g0[e]; A[kc][8 + e] = (_Float16)g1[e]; }
        }
        const float* w3 = args.w3[br]; const float* b3 = args.b3[br];
#pragma unroll
        for (int kc = 6; kc < 8; ++kc) {
#pragma unroll
            for (int e = 0; e < 16; ++e) {
                const int cc = (kc - 6) * 32 + ((e < 8) ? (e + 8 * half) : (16 + (e - 8) + 8 * half));
                float v = s2v[0] * w3[cc * 3 + 0] + s2v[1] * w3[cc * 3 + 1]
                        + s2v[2] * w3[cc * 3 + 2] + b3[cc];
                A[kc][e] = (_Float16)fmaxf(v, 0.0f);
            }
        }

        // Layer 1: [16x256] @ [256x64] + b1, relu -> LDS (A layout source for layer 2).
        const float* b1 = args.b1[br];
#pragma unroll
        for (int nt = 0; nt < 4; ++nt) {
            const float bval = b1[nt * 16 + nlo];
            v8f acc;
#pragma unroll
            for (int j = 0; j < 8; ++j) acc[j] = bval;
#pragma unroll
            for (int kc = 0; kc < 8; ++kc)
                acc = wmma_f16(A[kc], load_B(s_w1, lane, nt * 16, kc, 256), acc);
#pragma unroll
            for (int j = 0; j < 8; ++j)
                s_mid[w][(j + 8 * half) * 64 + nt * 16 + nlo] = (_Float16)fmaxf(acc[j], 0.0f);
        }

        // Layer 2: [16x64] @ [64x6] (N padded to 16 with zero columns).
        v16h A20 = load_A16x64(s_mid[w], lane, 0);
        v16h A21 = load_A16x64(s_mid[w], lane, 1);
        const float* w2 = args.w2[br]; const float* b2 = args.b2[br];
        v16h B20, B21;
#pragma unroll
        for (int e = 0; e < 16; ++e) { B20[e] = (_Float16)0.0f; B21[e] = (_Float16)0.0f; }
        if (nlo < 6) {
            const float* base = w2 + nlo * 64 + 8 * half;
            v8f g0 = *(const v8f*)(base);      v8f g1 = *(const v8f*)(base + 16);
            v8f g2 = *(const v8f*)(base + 32); v8f g3 = *(const v8f*)(base + 48);
#pragma unroll
            for (int e = 0; e < 8; ++e) {
                B20[e] = (_Float16)g0[e]; B20[8 + e] = (_Float16)g1[e];
                B21[e] = (_Float16)g2[e]; B21[8 + e] = (_Float16)g3[e];
            }
        }
        const float b2v = (nlo < 6) ? b2[nlo] : 0.0f;
        v8f acc2;
#pragma unroll
        for (int j = 0; j < 8; ++j) acc2[j] = b2v;
        acc2 = wmma_f16(A20, B20, acc2);
        acc2 = wmma_f16(A21, B21, acc2);

        if (br == 0) {
            if (nlo < 6) {
#pragma unroll
                for (int j = 0; j < 8; ++j)
                    args.out[(size_t)(bb + j + 8 * half) * 6 + nlo] = acc2[j];
            }
        } else {
            if (nlo < 6) {
#pragma unroll
                for (int j = 0; j < 8; ++j)
                    s_i6[w][(j + 8 * half) * 8 + nlo] = fmaxf(acc2[j], 0.0f);
            }
            __syncthreads();
            if (lane < 16) {
                const int r = bb + lane;
                float v[6]; float mx = -1e30f;
#pragma unroll
                for (int cc = 0; cc < 6; ++cc) { v[cc] = s_i6[w][lane * 8 + cc]; mx = fmaxf(mx, v[cc]); }
                float s = 0.0f;
#pragma unroll
                for (int cc = 0; cc < 6; ++cc) s += __expf(v[cc] - mx);
                const float lse = __logf(s);
#pragma unroll
                for (int cc = 0; cc < 6; ++cc) {
                    args.out[(size_t)B * 6     + (size_t)r * 6 + cc] = v[cc] - mx - lse;
                    args.out[(size_t)B * 6 * 2 + (size_t)r * 6 + cc] = v[cc];
                }
            }
        }
    }
}

extern "C" void kernel_launch(void* const* d_in, const int* in_sizes, int n_in,
                              void* d_out, int out_size, void* d_ws, size_t ws_size,
                              hipStream_t stream) {
    const int B = in_sizes[0] / TT;   // x0 is [B, T, 1]

    // Six LSTM runs: (x-index, param-set index among {q00,q01,i00,i01,i02}).
    const int rx[6] = {0, 1, 2, 0, 1, 2};
    const int rp[6] = {0, 1, 1, 2, 3, 4};

    LstmArgs la;
    for (int r = 0; r < 6; ++r) {
        la.x[r]   = (const float*)d_in[rx[r]];
        la.wih[r] = (const float*)d_in[4 + 3 * rp[r]];
        la.whh[r] = (const float*)d_in[5 + 3 * rp[r]];
        la.b[r]   = (const float*)d_in[6 + 3 * rp[r]];
    }
    la.hn = (float*)d_ws;
    la.B  = B;

    dim3 lgrid(B / 128, 6);
    lstm6_kernel<<<lgrid, 256, 0, stream>>>(la);

    HeadArgs ha;
    ha.s2    = (const float*)d_in[3];
    ha.hn    = (const float*)d_ws;
    ha.w3[0] = (const float*)d_in[19]; ha.b3[0] = (const float*)d_in[20];
    ha.w1[0] = (const float*)d_in[21]; ha.b1[0] = (const float*)d_in[22];
    ha.w2[0] = (const float*)d_in[23]; ha.b2[0] = (const float*)d_in[24];
    ha.w3[1] = (const float*)d_in[25]; ha.b3[1] = (const float*)d_in[26];
    ha.w1[1] = (const float*)d_in[27]; ha.b1[1] = (const float*)d_in[28];
    ha.w2[1] = (const float*)d_in[29]; ha.b2[1] = (const float*)d_in[30];
    ha.out   = (float*)d_out;
    ha.B     = B;

    head_kernel<<<dim3(B / 128), 256, 0, stream>>>(ha);
}
